// PointNet2FeatureExtractor_14946486190581
// MI455X (gfx1250) — matrix operand order repaired and checked by reading, not compiled, not run on hardware
//
#include <hip/hip_runtime.h>
#include <cstdint>
#include <cstddef>

// ---------------------------------------------------------------------------
// Types for CDNA5 WMMA (wave32, 16x16x32 f16 -> f32)
// ---------------------------------------------------------------------------
typedef _Float16 f16;
typedef _Float16 v8h  __attribute__((ext_vector_type(8)));
typedef _Float16 v16h __attribute__((ext_vector_type(16)));
typedef float    v8f  __attribute__((ext_vector_type(8)));

#define BATCH 16
#define KSAMP 16   // nsample for all SA layers
#define EPSBN 1e-5f

static inline int align32i(int x) { return (x + 31) & ~31; }

// ---------------------------------------------------------------------------
// Farthest point sampling: one workgroup per batch element.
// Emits fps index sequence (far BEFORE update, init far=0, matching the
// reference scan) and directly gathers new_xyz.
// ---------------------------------------------------------------------------
__global__ void fps_kernel(const float* __restrict__ xyz, int N, int npoint,
                           int* __restrict__ fps_idx, float* __restrict__ new_xyz) {
  __shared__ float dist[2048];
  __shared__ float rmax[256];
  __shared__ int   rarg[256];
  int b = blockIdx.x;
  int t = threadIdx.x;
  const float* X = xyz + (size_t)b * N * 3;
  for (int i = t; i < N; i += 256) dist[i] = 1e10f;
  __syncthreads();
  int far = 0;
  for (int it = 0; it < npoint; ++it) {
    float cx = X[far * 3 + 0];
    float cy = X[far * 3 + 1];
    float cz = X[far * 3 + 2];
    if (t == 0) {
      fps_idx[b * npoint + it] = far;
      float* nx = new_xyz + ((size_t)b * npoint + it) * 3;
      nx[0] = cx; nx[1] = cy; nx[2] = cz;
    }
    float best = -1.0f; int barg = 0;
    for (int i = t; i < N; i += 256) {
      float dx = X[i * 3 + 0] - cx;
      float dy = X[i * 3 + 1] - cy;
      float dz = X[i * 3 + 2] - cz;
      float d  = dx * dx + dy * dy + dz * dz;
      float dm = fminf(dist[i], d);
      dist[i] = dm;
      if (dm > best) { best = dm; barg = i; }
    }
    rmax[t] = best; rarg[t] = barg;
    __syncthreads();
    for (int s = 128; s > 0; s >>= 1) {
      if (t < s && rmax[t + s] > rmax[t]) { rmax[t] = rmax[t + s]; rarg[t] = rarg[t + s]; }
      __syncthreads();
    }
    far = rarg[0];
    __syncthreads();
  }
}

// ---------------------------------------------------------------------------
// Ball query: one thread per (b,s) query; first KSAMP points within radius,
// in index order, padded with the first hit (== reference sort semantics).
// ---------------------------------------------------------------------------
__global__ void ball_query_kernel(const float* __restrict__ xyz,
                                  const float* __restrict__ new_xyz,
                                  int N, int S, float r2, int* __restrict__ idx) {
  int i = blockIdx.x * 256 + threadIdx.x;
  if (i >= BATCH * S) return;
  int b = i / S;
  const float* X = xyz + (size_t)b * N * 3;
  float cx = new_xyz[(size_t)i * 3 + 0];
  float cy = new_xyz[(size_t)i * 3 + 1];
  float cz = new_xyz[(size_t)i * 3 + 2];
  int out[KSAMP];
  int cnt = 0, first = 0;
  for (int j = 0; j < N && cnt < KSAMP; ++j) {
    float dx = X[j * 3 + 0] - cx;
    float dy = X[j * 3 + 1] - cy;
    float dz = X[j * 3 + 2] - cz;
    float d = dx * dx + dy * dy + dz * dz;
    if (d <= r2) { if (cnt == 0) first = j; out[cnt++] = j; }
  }
  for (int k = cnt; k < KSAMP; ++k) out[k] = first;
  for (int k = 0; k < KSAMP; ++k) idx[(size_t)i * KSAMP + k] = out[k];
}

// ---------------------------------------------------------------------------
// SA grouping: build h[row, Kpad] (f16) = [xyz[idx]-center (3), points[idx] (C), 0 pad]
// row = ((b*S + s)*16 + k). 32 lanes cooperate per row -> coalesced stores.
// ---------------------------------------------------------------------------
__global__ void sa_group_kernel(const float* __restrict__ xyz,
                                const float* __restrict__ points,
                                const float* __restrict__ new_xyz,
                                const int* __restrict__ idx,
                                int S, int N, int C, int Kpad,
                                f16* __restrict__ h, int Rows) {
  long tid = (long)blockIdx.x * 256 + threadIdx.x;
  int row  = (int)(tid >> 5);
  int lane = (int)(tid & 31);
  if (row >= Rows) return;
  int k  = row & (KSAMP - 1);
  int bs = row >> 4;
  int b  = bs / S;
  int j  = idx[(size_t)bs * KSAMP + k];
  const float* pt = xyz + ((size_t)b * N + j) * 3;
  const float* ct = new_xyz + (size_t)bs * 3;
  float g0 = pt[0] - ct[0], g1 = pt[1] - ct[1], g2 = pt[2] - ct[2];
  const float* src = points + ((size_t)b * N + j) * C;
  f16* hr = h + (size_t)row * Kpad;
  for (int c = lane; c < Kpad; c += 32) {
    float v;
    if (c < 3)           v = (c == 0) ? g0 : ((c == 1) ? g1 : g2);
    else if (c < 3 + C)  v = src[c - 3];
    else                 v = 0.0f;
    hr[c] = (f16)v;
  }
}

// ---------------------------------------------------------------------------
// Pack weights (out,in) f32 -> (out, Kpad) f16, zero-padded in K.
// ---------------------------------------------------------------------------
__global__ void pack_w_kernel(const float* __restrict__ W, int Nc, int Kc, int Kpad,
                              f16* __restrict__ Wp) {
  int i = blockIdx.x * 256 + threadIdx.x;
  if (i >= Nc * Kpad) return;
  int n = i / Kpad, k = i % Kpad;
  Wp[i] = (k < Kc) ? (f16)W[(size_t)n * Kc + k] : (f16)0.0f;
}

// ---------------------------------------------------------------------------
// A-fragment loader for v_wmma_f32_16x16x32_f16 (16-bit A 16x32 layout):
// lane-half s reads contiguous 8-half runs at k + s*8 and k + 16 + s*8.
// ---------------------------------------------------------------------------
__device__ __forceinline__ v16h load_afrag(const f16* __restrict__ rowp, int k, int s) {
  v8h lo = *(const v8h*)(rowp + k + s * 8);
  v8h hi = *(const v8h*)(rowp + k + 16 + s * 8);
  return __builtin_shufflevector(lo, hi, 0, 1, 2, 3, 4, 5, 6, 7,
                                 8, 9, 10, 11, 12, 13, 14, 15);
}

// ---------------------------------------------------------------------------
// WMMA GEMM: Y[M,Nc] = A[M,Kpad](f16) * Wp[Nc,Kpad]^T(f16) + bias.
// Register-blocked: each wave owns a 32 x (NB*16) output block
// (2 M-tiles x NB N-tiles = 2*NB independent v8f accumulators), so each
// K-step issues 2 A loads + NB B loads feeding 2*NB WMMAs (~21 FLOP/B at
// NB=4) and the independent accumulator chains hide XDL latency.
// B fragment: column N=lane&15, 8-half runs at k + s*8 and k + 16 + s*8 of the
// row-major (Nc x Kpad) packed weight row.
// ---------------------------------------------------------------------------
template <int NB>
__global__ void wmma_gemm_bias_kernel(const f16* __restrict__ A,
                                      const f16* __restrict__ Wp,
                                      const float* __restrict__ bias,
                                      float* __restrict__ Y,
                                      int Mblocks, int Nblocks, int Kpad, int Ncols) {
  int wave = (int)(((long)blockIdx.x * 256 + threadIdx.x) >> 5);
  int lane = threadIdx.x & 31;
  if (wave >= Mblocks * Nblocks) return;
  int bm = wave / Nblocks;
  int bn = wave % Nblocks;
  int r  = lane & 15;
  int s  = lane >> 4;

  const f16* arow0 = A + (size_t)(bm * 32 + r) * Kpad;
  const f16* arow1 = arow0 + (size_t)16 * Kpad;
  const f16* brow[NB];
#pragma unroll
  for (int t = 0; t < NB; ++t)
    brow[t] = Wp + (size_t)((bn * NB + t) * 16 + r) * Kpad;

  v8f acc0[NB];
  v8f acc1[NB];
#pragma unroll
  for (int t = 0; t < NB; ++t) { acc0[t] = (v8f){}; acc1[t] = (v8f){}; }

  for (int k = 0; k < Kpad; k += 32) {
    __builtin_prefetch((const void*)(arow0 + k + 32), 0, 1);
    __builtin_prefetch((const void*)(arow1 + k + 32), 0, 1);
    v16h a0 = load_afrag(arow0, k, s);
    v16h a1 = load_afrag(arow1, k, s);
#pragma unroll
    for (int t = 0; t < NB; ++t) {
      v16h bm16 = __builtin_shufflevector(*(const v8h*)(brow[t] + k + s * 8),
                                          *(const v8h*)(brow[t] + k + 16 + s * 8),
                                          0, 1, 2, 3, 4, 5, 6, 7,
                                          8, 9, 10, 11, 12, 13, 14, 15);
      acc0[t] = __builtin_amdgcn_wmma_f32_16x16x32_f16(
          false, a0, false, bm16, (short)0, acc0[t], false, false);
      acc1[t] = __builtin_amdgcn_wmma_f32_16x16x32_f16(
          false, a1, false, bm16, (short)0, acc1[t], false, false);
    }
  }

#pragma unroll
  for (int t = 0; t < NB; ++t) {
    int col = (bn * NB + t) * 16 + r;
    float bv = bias[col];
    int row0 = bm * 32 + s * 8;
#pragma unroll
    for (int v = 0; v < 8; ++v)
      Y[(size_t)(row0 + v) * Ncols + col] = acc0[t][v] + bv;
    int row1 = bm * 32 + 16 + s * 8;
#pragma unroll
    for (int v = 0; v < 8; ++v)
      Y[(size_t)(row1 + v) * Ncols + col] = acc1[t][v] + bv;
  }
}

// ---------------------------------------------------------------------------
// Per-channel sum / sum-of-squares over all M rows (for BN mean/var).
// ---------------------------------------------------------------------------
__global__ void colstat_kernel(const float* __restrict__ Y, int M, int Nc,
                               float* __restrict__ sum, float* __restrict__ sumsq) {
  int r0 = blockIdx.x * 256;
  int rend = r0 + 256; if (rend > M) rend = M;
  for (int c = threadIdx.x; c < Nc; c += 256) {
    float s = 0.0f, ss = 0.0f;
    for (int r = r0; r < rend; ++r) {
      float v = Y[(size_t)r * Nc + c];
      s += v; ss += v * v;
    }
    atomicAdd(&sum[c], s);
    atomicAdd(&sumsq[c], ss);
  }
}

// ---------------------------------------------------------------------------
// BN (biased var) + ReLU. f16 variant feeds next GEMM staging (dense Nc),
// f32 variant produces points buffers / final output.
// ---------------------------------------------------------------------------
__global__ void bn_relu_f16_kernel(const float* __restrict__ Y,
                                   const float* __restrict__ sum,
                                   const float* __restrict__ sumsq,
                                   const float* __restrict__ g,
                                   const float* __restrict__ beta,
                                   int M, int Nc, f16* __restrict__ out) {
  long i = (long)blockIdx.x * 256 + threadIdx.x;
  if (i >= (long)M * Nc) return;
  int c = (int)(i % Nc);
  float inv = 1.0f / (float)M;
  float mean = sum[c] * inv;
  float var  = sumsq[c] * inv - mean * mean;
  float x = (Y[i] - mean) * rsqrtf(var + EPSBN) * g[c] + beta[c];
  out[i] = (f16)fmaxf(x, 0.0f);
}

__global__ void bn_relu_f32_kernel(const float* __restrict__ Y,
                                   const float* __restrict__ sum,
                                   const float* __restrict__ sumsq,
                                   const float* __restrict__ g,
                                   const float* __restrict__ beta,
                                   int M, int Nc, float* __restrict__ out) {
  long i = (long)blockIdx.x * 256 + threadIdx.x;
  if (i >= (long)M * Nc) return;
  int c = (int)(i % Nc);
  float inv = 1.0f / (float)M;
  float mean = sum[c] * inv;
  float var  = sumsq[c] * inv - mean * mean;
  float x = (Y[i] - mean) * rsqrtf(var + EPSBN) * g[c] + beta[c];
  out[i] = fmaxf(x, 0.0f);
}

// ---------------------------------------------------------------------------
// Max pool over the KSAMP group dimension: (BS*16, Nc) f16 -> (BS, Nc) f32.
// ---------------------------------------------------------------------------
__global__ void maxpool_kernel(const f16* __restrict__ h, int BS, int Nc,
                               float* __restrict__ out) {
  int i = blockIdx.x * 256 + threadIdx.x;
  if (i >= BS * Nc) return;
  int bs = i / Nc, c = i % Nc;
  const f16* base = h + (size_t)bs * KSAMP * Nc + c;
  float m = -3.0e38f;
#pragma unroll
  for (int k = 0; k < KSAMP; ++k) m = fmaxf(m, (float)base[(size_t)k * Nc]);
  out[i] = m;
}

// ---------------------------------------------------------------------------
// 3-NN search + inverse-distance weights (reference: w = 1/(d+1e-8), normed).
// ---------------------------------------------------------------------------
__global__ void three_nn_kernel(const float* __restrict__ xyz1,
                                const float* __restrict__ xyz2,
                                int N, int S2,
                                int* __restrict__ nidx, float* __restrict__ nw) {
  int i = blockIdx.x * 256 + threadIdx.x;
  if (i >= BATCH * N) return;
  int b = i / N;
  float px = xyz1[(size_t)i * 3 + 0];
  float py = xyz1[(size_t)i * 3 + 1];
  float pz = xyz1[(size_t)i * 3 + 2];
  const float* X2 = xyz2 + (size_t)b * S2 * 3;
  float d0 = 1e30f, d1 = 1e30f, d2 = 1e30f;
  int   i0 = 0,     i1 = 0,     i2 = 0;
  for (int j = 0; j < S2; ++j) {
    float dx = X2[j * 3 + 0] - px;
    float dy = X2[j * 3 + 1] - py;
    float dz = X2[j * 3 + 2] - pz;
    float d = dx * dx + dy * dy + dz * dz;
    if (d < d0)      { d2 = d1; i2 = i1; d1 = d0; i1 = i0; d0 = d; i0 = j; }
    else if (d < d1) { d2 = d1; i2 = i1; d1 = d;  i1 = j; }
    else if (d < d2) { d2 = d;  i2 = j; }
  }
  float w0 = 1.0f / (d0 + 1e-8f);
  float w1 = 1.0f / (d1 + 1e-8f);
  float w2 = 1.0f / (d2 + 1e-8f);
  float ws = 1.0f / (w0 + w1 + w2);
  nidx[(size_t)i * 3 + 0] = i0;
  nidx[(size_t)i * 3 + 1] = i1;
  nidx[(size_t)i * 3 + 2] = i2;
  nw[(size_t)i * 3 + 0] = w0 * ws;
  nw[(size_t)i * 3 + 1] = w1 * ws;
  nw[(size_t)i * 3 + 2] = w2 * ws;
}

// ---------------------------------------------------------------------------
// FP concat: h[row, Kpad] = [points1 (C1), interp(points2 via 3NN) (C2), 0].
// ---------------------------------------------------------------------------
__global__ void fp_concat_kernel(const float* __restrict__ points1, int C1,
                                 const float* __restrict__ points2, int C2, int S2,
                                 const int* __restrict__ nidx,
                                 const float* __restrict__ nw,
                                 int N, int Kpad, f16* __restrict__ h, int Rows) {
  long tid = (long)blockIdx.x * 256 + threadIdx.x;
  int row  = (int)(tid >> 5);
  int lane = (int)(tid & 31);
  if (row >= Rows) return;
  int b = row / N;
  int j0 = nidx[(size_t)row * 3 + 0];
  int j1 = nidx[(size_t)row * 3 + 1];
  int j2 = nidx[(size_t)row * 3 + 2];
  float w0 = nw[(size_t)row * 3 + 0];
  float w1 = nw[(size_t)row * 3 + 1];
  float w2 = nw[(size_t)row * 3 + 2];
  const float* p2b = points2 + (size_t)b * S2 * C2;
  f16* hr = h + (size_t)row * Kpad;
  for (int c = lane; c < Kpad; c += 32) {
    float v;
    if (c < C1)            v = points1[(size_t)row * C1 + c];
    else if (c < C1 + C2) {
      int cc = c - C1;
      v = w0 * p2b[(size_t)j0 * C2 + cc]
        + w1 * p2b[(size_t)j1 * C2 + cc]
        + w2 * p2b[(size_t)j2 * C2 + cc];
    } else                 v = 0.0f;
    hr[c] = (f16)v;
  }
}

// ---------------------------------------------------------------------------
// Host-side orchestration
// ---------------------------------------------------------------------------
struct LayerP { const float *W, *b, *g, *beta; int out, in; };

extern "C" void kernel_launch(void* const* d_in, const int* in_sizes, int n_in,
                              void* d_out, int out_size, void* d_ws, size_t ws_size,
                              hipStream_t stream) {
  (void)in_sizes; (void)n_in; (void)out_size; (void)ws_size;

  const float* xyz = (const float*)d_in[0];

  // ---- parameter table (setup_inputs dict order: xyz, then params sa->fp,
  //      each layer W, b, g, beta) ----
  static const int sa_inC[4]     = {6, 67, 131, 259};
  static const int sa_outC[4][3] = {{32, 32, 64}, {64, 64, 128},
                                    {128, 128, 256}, {256, 256, 512}};
  static const int fp_inC[4]     = {768, 384, 320, 131};
  static const int fp_outC[4][2] = {{256, 256}, {256, 256}, {256, 128}, {128, 128}};
  static const float radius2[4]  = {0.05f * 0.05f, 0.1f * 0.1f, 0.2f * 0.2f, 0.4f * 0.4f};
  static const int Nl[5] = {2048, 1024, 512, 256, 128};
  static const int Cl_[5] = {3, 64, 128, 256, 512};

  LayerP sa[4][3], fp[4][2];
  int pi = 1;
  for (int l = 0; l < 4; ++l)
    for (int j = 0; j < 3; ++j) {
      sa[l][j].W = (const float*)d_in[pi++];
      sa[l][j].b = (const float*)d_in[pi++];
      sa[l][j].g = (const float*)d_in[pi++];
      sa[l][j].beta = (const float*)d_in[pi++];
      sa[l][j].out = sa_outC[l][j];
      sa[l][j].in  = (j == 0) ? sa_inC[l] : sa_outC[l][j - 1];
    }
  for (int l = 0; l < 4; ++l)
    for (int j = 0; j < 2; ++j) {
      fp[l][j].W = (const float*)d_in[pi++];
      fp[l][j].b = (const float*)d_in[pi++];
      fp[l][j].g = (const float*)d_in[pi++];
      fp[l][j].beta = (const float*)d_in[pi++];
      fp[l][j].out = fp_outC[l][j];
      fp[l][j].in  = (j == 0) ? fp_inC[l] : fp_outC[l][j - 1];
    }

  // ---- deterministic workspace bump allocator ----
  char* cur = (char*)d_ws;
  auto walloc = [&](size_t bytes) -> void* {
    void* r = (void*)cur;
    cur += (bytes + 255) & ~(size_t)255;
    return r;
  };
  float* x_lvl[5]; float* p_lvl[5];
  x_lvl[0] = (float*)xyz; p_lvl[0] = (float*)xyz;
  for (int l = 1; l <= 4; ++l) {
    x_lvl[l] = (float*)walloc((size_t)BATCH * Nl[l] * 3 * sizeof(float));
    p_lvl[l] = (float*)walloc((size_t)BATCH * Nl[l] * Cl_[l] * sizeof(float));
  }
  float* fpo[3];
  fpo[0] = (float*)walloc((size_t)BATCH * 256 * 256 * sizeof(float));
  fpo[1] = (float*)walloc((size_t)BATCH * 512 * 256 * sizeof(float));
  fpo[2] = (float*)walloc((size_t)BATCH * 1024 * 128 * sizeof(float));

  const size_t MAXE = 16777216;  // max rows*Kpad over all layers
  f16*   HA    = (f16*)walloc(MAXE * sizeof(f16));
  f16*   HB    = (f16*)walloc(MAXE * sizeof(f16));
  float* Ybuf  = (float*)walloc(MAXE * sizeof(float));
  f16*   Wp    = (f16*)walloc(262144 * sizeof(f16));
  float* stats = (float*)walloc(1024 * sizeof(float));  // [sum(512), sumsq(512)]
  int*   fpsI  = (int*)walloc((size_t)BATCH * 1024 * sizeof(int));
  int*   bidx  = (int*)walloc((size_t)BATCH * 1024 * KSAMP * sizeof(int));
  int*   nidx  = (int*)walloc((size_t)BATCH * 2048 * 3 * sizeof(int));
  float* nwb   = (float*)walloc((size_t)BATCH * 2048 * 3 * sizeof(float));

  // ---- one conv layer: GEMM + BN stats + BN/ReLU ----
  auto conv_bn = [&](const f16* hin, int M, int Kpad, const LayerP& L,
                     void* outPtr, bool outF32) {
    int wtot = L.out * Kpad;
    pack_w_kernel<<<dim3((wtot + 255) / 256), dim3(256), 0, stream>>>(
        L.W, L.out, L.in, Kpad, Wp);
    int mt = M / 16, nt = L.out / 16;          // mt even for all layers
    int mblocks = mt / 2;
    int nb = (nt % 4 == 0) ? 4 : 2;
    int nblocks = nt / nb;
    long waves = (long)mblocks * nblocks;
    dim3 grid((unsigned)((waves * 32 + 255) / 256));
    if (nb == 4)
      wmma_gemm_bias_kernel<4><<<grid, dim3(256), 0, stream>>>(
          hin, Wp, L.b, Ybuf, mblocks, nblocks, Kpad, L.out);
    else
      wmma_gemm_bias_kernel<2><<<grid, dim3(256), 0, stream>>>(
          hin, Wp, L.b, Ybuf, mblocks, nblocks, Kpad, L.out);
    hipMemsetAsync(stats, 0, 1024 * sizeof(float), stream);
    colstat_kernel<<<dim3((M + 255) / 256), dim3(256), 0, stream>>>(
        Ybuf, M, L.out, stats, stats + 512);
    long tot = (long)M * L.out;
    if (outF32)
      bn_relu_f32_kernel<<<dim3((unsigned)((tot + 255) / 256)), dim3(256), 0, stream>>>(
          Ybuf, stats, stats + 512, L.g, L.beta, M, L.out, (float*)outPtr);
    else
      bn_relu_f16_kernel<<<dim3((unsigned)((tot + 255) / 256)), dim3(256), 0, stream>>>(
          Ybuf, stats, stats + 512, L.g, L.beta, M, L.out, (f16*)outPtr);
  };

  // ======================= SA layers =======================
  for (int l = 0; l < 4; ++l) {
    int N = Nl[l], S = Nl[l + 1], C = Cl_[l];
    fps_kernel<<<dim3(BATCH), dim3(256), 0, stream>>>(x_lvl[l], N, S, fpsI, x_lvl[l + 1]);
    ball_query_kernel<<<dim3((BATCH * S + 255) / 256), dim3(256), 0, stream>>>(
        x_lvl[l], x_lvl[l + 1], N, S, radius2[l], bidx);
    int Rows  = BATCH * S * KSAMP;
    int Kpad0 = align32i(C + 3);
    sa_group_kernel<<<dim3(Rows / 8), dim3(256), 0, stream>>>(
        x_lvl[l], p_lvl[l], x_lvl[l + 1], bidx, S, N, C, Kpad0, HA, Rows);
    f16* hin = HA; f16* hout = HB;
    int Kp = Kpad0;
    for (int j = 0; j < 3; ++j) {
      conv_bn(hin, Rows, Kp, sa[l][j], hout, /*outF32=*/false);
      f16* tmp = hin; hin = hout; hout = tmp;
      Kp = sa[l][j].out;  // out channels are multiples of 32 -> next Kpad dense
    }
    int outC = sa[l][2].out;
    int BS = BATCH * S;
    maxpool_kernel<<<dim3((BS * outC + 255) / 256), dim3(256), 0, stream>>>(
        hin, BS, outC, p_lvl[l + 1]);
  }

  // ======================= FP layers =======================
  const float* curp = p_lvl[4];
  int curC2 = 512;
  for (int i = 0; i < 4; ++i) {
    const float* xyz1 = x_lvl[3 - i]; int N1 = Nl[3 - i];
    const float* xyz2 = x_lvl[4 - i]; int S2 = Nl[4 - i];
    const float* pts1 = p_lvl[3 - i]; int C1 = Cl_[3 - i];
    three_nn_kernel<<<dim3((BATCH * N1 + 255) / 256), dim3(256), 0, stream>>>(
        xyz1, xyz2, N1, S2, nidx, nwb);
    int Rows = BATCH * N1;
    int Kpad = align32i(C1 + curC2);
    fp_concat_kernel<<<dim3(Rows / 8), dim3(256), 0, stream>>>(
        pts1, C1, curp, curC2, S2, nidx, nwb, N1, Kpad, HA, Rows);
    conv_bn(HA, Rows, Kpad, fp[i][0], HB, /*outF32=*/false);
    float* outp = (i == 3) ? (float*)d_out : fpo[i];
    conv_bn(HB, Rows, fp[i][0].out, fp[i][1], outp, /*outF32=*/true);
    curp = outp;
    curC2 = fp[i][1].out;
  }
}
